// RouterAugmentedLinear_20177756357134
// MI455X (gfx1250) — matrix operand order repaired and checked by reading, not compiled, not run on hardware
//
#include <hip/hip_runtime.h>
#include <stdint.h>

// ---------------------------------------------------------------------------
// RouterAugmentedLinear forward for MI455X (gfx1250), wave32 + WMMA bf16.
//   router = x @ Wr^T + br          (bf16 WMMA, f32 accum)  -> ws
//   thr[n] = 1024th-largest(router[n,:])  (bit-radix select in LDS)
//   out    = (x @ Wo^T + bo) * (router >= thr)
// GEMM stages operands global->LDS with GLOBAL_LOAD_ASYNC_TO_LDS_B128
// (ASYNCcnt-tracked, double-buffered) and feeds WMMA fragments from LDS.
// Wave tile 32x64: 8 WMMA per 12 ds_load_b128 per k-step.
// ---------------------------------------------------------------------------

typedef __attribute__((ext_vector_type(16))) __bf16 v16bf;
typedef __attribute__((ext_vector_type(8)))  float  v8f;

#ifndef USE_ASYNC_COPY
#define USE_ASYNC_COPY 1
#endif

// order-preserving f32 -> u32 key (ascending)
__device__ __forceinline__ uint32_t f32key(float f) {
  uint32_t u = __float_as_uint(f);
  return (u & 0x80000000u) ? ~u : (u | 0x80000000u);
}

__device__ __forceinline__ uint16_t f32_to_bf16_rne(float f) {
  uint32_t u = __float_as_uint(f);
  return (uint16_t)((u + 0x7FFFu + ((u >> 16) & 1u)) >> 16);
}

// 16B global -> LDS copy. Async path: per-lane DMA, tracked by ASYNCcnt.
// Generic LDS pointer truncated to 32 bits == LDS byte address (ISA 10.2).
__device__ __forceinline__ void cp16(uint16_t* l, const uint16_t* g) {
#if USE_ASYNC_COPY
  asm volatile("global_load_async_to_lds_b128 %0, %1, off"
               :: "v"((uint32_t)(uintptr_t)l), "v"((uint64_t)(uintptr_t)g)
               : "memory");
#else
  *(uint4*)l = *(const uint4*)g;
#endif
}

#if USE_ASYNC_COPY
#define WAIT_ASYNC(n) asm volatile("s_wait_asynccnt " #n ::: "memory")
#else
#define WAIT_ASYNC(n)
#endif

// ---------------------------------------------------------------------------
// Kernel 1: fp32 -> bf16 convert (vectorized x4)
// ---------------------------------------------------------------------------
__global__ __launch_bounds__(256) void cvt_bf16_kernel(
    const float* __restrict__ src, uint16_t* __restrict__ dst, long n4) {
  long i = blockIdx.x * (long)blockDim.x + threadIdx.x;
  const long stride = (long)gridDim.x * blockDim.x;
  const float4* s4 = (const float4*)src;
  ushort4* d4 = (ushort4*)dst;
  for (; i < n4; i += stride) {
    float4 f = s4[i];
    ushort4 o;
    o.x = f32_to_bf16_rne(f.x);
    o.y = f32_to_bf16_rne(f.y);
    o.z = f32_to_bf16_rne(f.z);
    o.w = f32_to_bf16_rne(f.w);
    d4[i] = o;
  }
}

// ---------------------------------------------------------------------------
// Kernels 2 & 4: bf16 WMMA GEMM  out[n,o] = sum_d A[n,d]*B[o,d] + bias[o]
// Block tile 64(M) x 128(N), 128 threads = 4 waves (2M x 2N), wave 32x64.
// K staged 64 at a time into double-buffered LDS via async copies.
// MODE 0: write raw f32 (router pass). MODE 1: apply top-k mask -> d_out.
// ---------------------------------------------------------------------------
#define BM 64
#define BN 128
#define KST 64
#define AST 72                         // padded row stride (halves): 64+8
#define A_HALVES (BM * AST)            // 4608
#define B_HALVES (BN * AST)            // 9216
#define STG (A_HALVES + B_HALVES)      // 13824 halves = 27,648 B per stage
#define NCP 12                         // 16B copies per thread per stage

template <int MODE>
__global__ __launch_bounds__(128) void gemm_bf16_wmma_kernel(
    const uint16_t* __restrict__ A16,   // [N, D] bf16
    const uint16_t* __restrict__ B16,   // [O, D] bf16
    const float* __restrict__ bias,     // [O]
    const float* __restrict__ router,   // [N, O]  (MODE==1)
    const uint32_t* __restrict__ uth,   // [N]     (MODE==1)
    float* __restrict__ out,            // [N, O]
    int D, int O) {
  __shared__ uint16_t lds[2 * STG];    // 55,296 B

  const int tid  = threadIdx.x;
  const int lane = tid & 31;
  const int wv   = tid >> 5;
  const int l15  = lane & 15;
  const int hi   = lane >> 4;
  const int mblk = blockIdx.y * BM;
  const int nblk = blockIdx.x * BN;
  const int mw   = (wv & 1) * 32;      // wave M offset in block tile
  const int nw   = (wv >> 1) * 64;     // wave N offset in block tile

  // Per-thread copy assignments: 12 x 16B chunks per stage (4 A + 8 B).
  // chunk id c: row = c>>3, col-halves = (c&7)*8
  const uint16_t* gsrc[NCP];
  uint32_t loff[NCP];                  // LDS half-offset within a stage
#pragma unroll
  for (int j = 0; j < 4; ++j) {        // A: 64 rows x 8 chunks = 512
    int c = tid + 128 * j;
    int row = c >> 3, cc = (c & 7) * 8;
    gsrc[j] = A16 + (size_t)(mblk + row) * (size_t)D + cc;
    loff[j] = row * AST + cc;
  }
#pragma unroll
  for (int j = 0; j < 8; ++j) {        // B: 128 rows x 8 chunks = 1024
    int c = tid + 128 * j;
    int row = c >> 3, cc = (c & 7) * 8;
    gsrc[4 + j] = B16 + (size_t)(nblk + row) * (size_t)D + cc;
    loff[4 + j] = A_HALVES + row * AST + cc;
  }

  v8f acc[2][4];
#pragma unroll
  for (int im = 0; im < 2; ++im)
#pragma unroll
    for (int in = 0; in < 4; ++in) acc[im][in] = (v8f){};

  const int S = D / KST;

  // prologue: stage 0 into buffer 0
#pragma unroll
  for (int j = 0; j < NCP; ++j) cp16(lds + loff[j], gsrc[j]);

  for (int s = 0; s < S; ++s) {
    if (s + 1 < S) {
      const int kb = (s + 1) * KST;
      uint16_t* buf = lds + ((s + 1) & 1) * STG;
#pragma unroll
      for (int j = 0; j < NCP; ++j) cp16(buf + loff[j], gsrc[j] + kb);
      WAIT_ASYNC(12);                  // stage s copies complete (own wave)
    } else {
      WAIT_ASYNC(0);
    }
    __syncthreads();                   // stage s visible to all waves

    const uint16_t* As = lds + (s & 1) * STG;
    const uint16_t* Bs = As + A_HALVES;
#pragma unroll
    for (int kk = 0; kk < KST; kk += 32) {
      union { v16bf v; uint4 q[2]; } a[2], b[4];
      // A frag (16x32): lane<16 k={0..7,16..23}, lane>=16 k={8..15,24..31}
#pragma unroll
      for (int im = 0; im < 2; ++im) {
        const uint16_t* pa = As + (mw + im * 16 + l15) * AST + kk + hi * 8;
        a[im].q[0] = *(const uint4*)pa;
        a[im].q[1] = *(const uint4*)(pa + 16);
      }
      // B frag (32x16): lane<16 k={0..7,16..23}, lane>=16 k={8..15,24..31}
#pragma unroll
      for (int in = 0; in < 4; ++in) {
        const uint16_t* pb = Bs + (nw + in * 16 + l15) * AST + kk + hi * 8;
        b[in].q[0] = *(const uint4*)pb;
        b[in].q[1] = *(const uint4*)(pb + 16);
      }
#pragma unroll
      for (int im = 0; im < 2; ++im)
#pragma unroll
        for (int in = 0; in < 4; ++in)
          acc[im][in] = __builtin_amdgcn_wmma_f32_16x16x32_bf16(
              false, a[im].v, false, b[in].v, (short)0, acc[im][in],
              false, false);
    }
    __syncthreads();                   // done reading buffer (s&1)
  }

  // C/D layout: lane l, vgpr v -> (m = v + 8*(l>>4), n = l&15)
  const int m0 = mblk + mw;
  const int n0 = nblk + nw;
#pragma unroll
  for (int im = 0; im < 2; ++im) {
#pragma unroll
    for (int in = 0; in < 4; ++in) {
      const int col = n0 + in * 16 + l15;
      const float bs = bias[col];
      const int mbase = m0 + im * 16 + hi * 8;
#pragma unroll
      for (int v = 0; v < 8; ++v) {
        const size_t idx = (size_t)(mbase + v) * (size_t)O + col;
        const float val = acc[im][in][v] + bs;
        if (MODE == 0) {
          out[idx] = val;
        } else {
          const uint32_t key = f32key(router[idx]);
          out[idx] = (key >= uth[mbase + v]) ? val : 0.0f;
        }
      }
    }
  }
}

// ---------------------------------------------------------------------------
// Kernel 3: per-token threshold = k-th largest router value.
// Row keys live in LDS; 32-step MSB->LSB prefix radix select (exact).
// ---------------------------------------------------------------------------
__global__ __launch_bounds__(256) void topk_thresh_kernel(
    const float* __restrict__ router, uint32_t* __restrict__ uth, int O, int k) {
  extern __shared__ uint32_t s_keys[];
  __shared__ int s_cnt;
  const int tid = threadIdx.x;
  const float* row = router + (size_t)blockIdx.x * (size_t)O;
  for (int i = tid; i < O; i += blockDim.x) s_keys[i] = f32key(row[i]);
  __syncthreads();

  uint32_t prefix = 0;
  for (int bit = 31; bit >= 0; --bit) {
    const uint32_t trial = prefix | (1u << bit);
    int c = 0;
    for (int i = tid; i < O; i += blockDim.x) c += (s_keys[i] >= trial) ? 1 : 0;
    __syncthreads();
    if (tid == 0) s_cnt = 0;
    __syncthreads();
    atomicAdd(&s_cnt, c);
    __syncthreads();
    if (s_cnt >= k) prefix = trial;    // keep bit: still >= k keys above
  }
  if (tid == 0) uth[blockIdx.x] = prefix;
}

// ---------------------------------------------------------------------------
// Host launcher
// ---------------------------------------------------------------------------
extern "C" void kernel_launch(void* const* d_in, const int* in_sizes, int n_in,
                              void* d_out, int out_size, void* d_ws, size_t ws_size,
                              hipStream_t stream) {
  const float* x  = (const float*)d_in[0];   // [B,S,D] = [N,D]
  const float* Wr = (const float*)d_in[1];   // [O,D]
  const float* br = (const float*)d_in[2];   // [O]
  const float* Wo = (const float*)d_in[3];   // [O,D]
  const float* bo = (const float*)d_in[4];   // [O]

  const long O = in_sizes[2];
  const long D = in_sizes[1] / O;
  const long N = in_sizes[0] / D;
  int k = (int)(O / 4); if (k < 1) k = 1;    // TOP_K_PERCENT = 0.25

  // workspace carve-out (256B aligned)
  char* ws = (char*)d_ws;
  size_t off = 0;
  auto carve = [&](size_t bytes) {
    void* p = ws + off;
    off = (off + bytes + 255) & ~(size_t)255;
    return p;
  };
  uint16_t* x16  = (uint16_t*)carve((size_t)N * D * 2);
  uint16_t* wr16 = (uint16_t*)carve((size_t)O * D * 2);
  uint16_t* wo16 = (uint16_t*)carve((size_t)O * D * 2);
  float*    rout = (float*)   carve((size_t)N * O * 4);
  uint32_t* uth  = (uint32_t*)carve((size_t)N * 4);
  (void)ws_size; (void)n_in; (void)out_size;

  // 1) fp32 -> bf16
  {
    long n4 = (N * D) / 4;
    int g = (int)((n4 + 255) / 256); if (g > 32768) g = 32768;
    cvt_bf16_kernel<<<g, 256, 0, stream>>>(x, x16, n4);
    n4 = (O * D) / 4;
    g = (int)((n4 + 255) / 256); if (g > 32768) g = 32768;
    cvt_bf16_kernel<<<g, 256, 0, stream>>>(Wr, wr16, n4);
    cvt_bf16_kernel<<<g, 256, 0, stream>>>(Wo, wo16, n4);
  }

  dim3 gemm_grid((unsigned)(O / BN), (unsigned)(N / BM));

  // 2) router GEMM
  gemm_bf16_wmma_kernel<0><<<gemm_grid, 128, 0, stream>>>(
      x16, wr16, br, nullptr, nullptr, rout, (int)D, (int)O);

  // 3) per-token top-k threshold
  topk_thresh_kernel<<<(unsigned)N, 256, (size_t)O * 4, stream>>>(
      rout, uth, (int)O, k);

  // 4) orig GEMM + mask -> d_out
  gemm_bf16_wmma_kernel<1><<<gemm_grid, 128, 0, stream>>>(
      x16, wo16, bo, rout, uth, (float*)d_out, (int)D, (int)O);
}
